// EncoderSARNN_86904368267555
// MI455X (gfx1250) — matrix operand order repaired and checked by reading, not compiled, not run on hardware
//
#include <hip/hip_runtime.h>
#include <hip/hip_bf16.h>
#include <math.h>

typedef _Float16 v16h __attribute__((ext_vector_type(16)));
typedef _Float16 v8h  __attribute__((ext_vector_type(8)));
typedef float    v8f  __attribute__((ext_vector_type(8)));

#define T_      512
#define B_      256
#define IDIM_   256
#define HDIM_   256
#define SSZ_    64
#define SDIM_   64
#define SDEPTH_ 4

// ---- weight-fragment table (f16, WMMA B-operand layout) in workspace ----
// One fragment = 32 lanes x 16 halves = 512 halves = 1 KB.
// B-layout (16-bit, K=32): lanes 0-15 hold N=lane, lanes 16-31 hold N=lane-16;
// half i -> K = (i<8 ? i : i+8) + 8*hiHalf (matches A).        [ISA 7.12.2/7.12.4]
#define FRAG_H  512
#define OFF_WEH 0                               // 8 kt x 16 nt
#define N_WEH   128
#define OFF_WR  (OFF_WEH + N_WEH*FRAG_H)        // [W_hh;W_sh] 16 kt x 16 nt
#define N_WR    256
#define OFF_WHS (OFF_WR + N_WR*FRAG_H)          // 8 kt x 4 nt
#define N_WHS   32
#define OFF_WHA (OFF_WHS + N_WHS*FRAG_H)        // 8 kt x 1 nt (cols 3..15 = 0)
#define N_WHA   8
#define MX_OFF_BYTES (512*1024)                 // f32 mx in C-fragment layout

static __device__ __forceinline__ v8f wmma_f16(v16h a, v16h b, v8f c) {
    return __builtin_amdgcn_wmma_f32_16x16x32_f16(false, a, false, b,
                                                  (short)0, c, false, false);
}

// A-operand fragment from a row-major f16 row in LDS.
// A-layout (16-bit, 16x32): lanes0-15 M=lane, halves i<8 -> K=i, i>=8 -> K=i+8;
// lanes16-31 M=lane-16, K offset +8.  => two ds_load_b128 per fragment.
static __device__ __forceinline__ v16h load_a_frag(const _Float16* rowp, int hi) {
    const _Float16* p = rowp + (hi ? 8 : 0);
    v8h a0 = *(const v8h*)(p);
    v8h a1 = *(const v8h*)(p + 16);
    return __builtin_shufflevector(a0, a1, 0,1,2,3,4,5,6,7,8,9,10,11,12,13,14,15);
}

// mx C-fragment address: mxf[((tile*16 + nt)*32 + lane)*8 + r]
static __device__ __forceinline__ size_t mxf_off(size_t tile, int nt, int lane) {
    return ((tile * 16 + (size_t)nt) * 32 + (size_t)lane) * 8;
}

// ---------------- weight repack: f32 row-major -> f16 B-fragments ----------------
__global__ __launch_bounds__(32) void repack_weights(
    const float* __restrict__ W_eh, const float* __restrict__ W_hh,
    const float* __restrict__ W_sh, const float* __restrict__ W_hs,
    const float* __restrict__ W_ha, _Float16* __restrict__ frags)
{
    int f = blockIdx.x;
    int lane = threadIdx.x;
    int hi = lane >> 4, nn = lane & 15;
    const float* W; int kt, nt, ncols, base;
    if (f < N_WEH) {
        W = W_eh; kt = f >> 4; nt = f & 15; ncols = HDIM_; base = OFF_WEH + f*FRAG_H;
    } else if (f < N_WEH + N_WR) {
        int g = f - N_WEH; kt = g >> 4; nt = g & 15; ncols = HDIM_;
        base = OFF_WR + g*FRAG_H;
        if (kt < 8) W = W_hh; else { W = W_sh; kt -= 8; }
    } else if (f < N_WEH + N_WR + N_WHS) {
        int g = f - (N_WEH + N_WR); kt = g >> 2; nt = g & 3; ncols = SDIM_;
        W = W_hs; base = OFF_WHS + g*FRAG_H;
    } else {
        int g = f - (N_WEH + N_WR + N_WHS); kt = g; nt = 0; ncols = 3;
        W = W_ha; base = OFF_WHA + g*FRAG_H;
    }
    int n = nt*16 + nn;
    v16h out;
    for (int i = 0; i < 16; ++i) {
        int k = kt*32 + (i & 7) + ((i >> 3) << 4) + (hi ? 8 : 0); // B-operand K mapping
        float v = (n < ncols) ? W[(size_t)k*ncols + n] : 0.0f;
        out[i] = (_Float16)v;
    }
    *(v16h*)(frags + base + lane*16) = out;
}

// -- parallel precompute: mx = embs @ W_eh + (b_eh+b_hh+b_sh), C-fragment layout --
__global__ __launch_bounds__(256) void precompute_mx(
    const float* __restrict__ embs, const float* __restrict__ b_eh,
    const float* __restrict__ b_hh, const float* __restrict__ b_sh,
    const _Float16* __restrict__ frags, float* __restrict__ mxf)
{
    __shared__ __align__(16) _Float16 at[16*256];
    const size_t tile = blockIdx.x;                      // 16 rows of flattened (t*B+b)
    const size_t row0 = tile * 16;
    const int tid  = threadIdx.x;
    for (int e = tid; e < 16*256; e += 256)
        at[e] = (_Float16)embs[row0*256 + e];
    __syncthreads();

    const int wave = tid >> 5, lane = tid & 31;
    const int nn = lane & 15, hi = lane >> 4;
    const _Float16* hrow = at + (size_t)(lane & 15)*256;
    for (int w = 0; w < 2; ++w) {
        int nt = wave*2 + w;
        v8f c = {};                                      // zero init: no VALU in chain
        for (int kt = 0; kt < 8; ++kt) {
            v16h a = load_a_frag(hrow + kt*32, hi);
            v16h b = *(const v16h*)(frags + OFF_WEH + (size_t)(kt*16+nt)*FRAG_H + lane*16);
            c = wmma_f16(a, b, c);
        }
        float bias = b_eh[nt*16+nn] + b_hh[nt*16+nn] + b_sh[nt*16+nn];
        for (int r = 0; r < 8; ++r) c[r] += bias;        // fold bias post-chain
        *(v8f*)(mxf + mxf_off(tile, nt, lane)) = c;      // coalesced fragment store
    }
}

// ---------------- persistent recurrent kernel: 16 WGs, T=512 steps ----------------
__global__ __launch_bounds__(256) void sarnn_serial(
    const float* __restrict__ mem_bias, const float* __restrict__ b_hs,
    const float* __restrict__ b_ha, const _Float16* __restrict__ frags,
    const float* __restrict__ mxf, float* __restrict__ out)
{
    __shared__ __align__(16) _Float16 hid[16*256];       // 8 KB
    __shared__ __align__(16) _Float16 stk[2][16*4096];   // 2 x 128 KB ping-pong
    __shared__ __align__(16) _Float16 pushv[16*64];      // 2 KB
    __shared__ __align__(16) _Float16 botv[64];
    __shared__ float pvals[16][3];
    __shared__ float logits[16][4];

    const int tid  = threadIdx.x;
    const int m0   = blockIdx.x * 16;                    // batch tile
    const int wave = tid >> 5, lane = tid & 31;
    const int hi   = lane >> 4, nn = lane & 15, mBase = hi ? 8 : 0;

    // biases hoisted out of the t-loop (scalar per lane)
    const float bhs0 = (wave < 4) ? b_hs[wave*16 + nn] : 0.f;
    const float bha0 = (nn < 3) ? b_ha[nn] : 0.f;

    for (int e = tid; e < 16*256; e += 256) hid[e] = (_Float16)0.f;
    for (int e = tid; e < 64;     e += 256) botv[e] = (_Float16)mem_bias[e];
    for (int e = tid; e < 16*4096; e += 256) stk[0][e] = (_Float16)mem_bias[e & 63];
    __syncthreads();

    int cur = 0;
    for (int t = 0; t < T_; ++t) {
        const size_t outbase = ((size_t)t*B_ + m0) * 256;
        const size_t tile    = (size_t)t*16 + blockIdx.x;    // B_/16 tiles per step
        const _Float16* hrow = hid + (size_t)(lane & 15)*256;
        const _Float16* srow = &stk[cur][(size_t)(lane & 15)*4096]; // tops = first 256 halves

        // prefetch next step's mx fragments (only HBM stream on the serial path)
        if (t + 1 < T_) {
            size_t ntile = tile + 16;
            __builtin_prefetch(mxf + mxf_off(ntile, wave*2,     lane), 0, 0);
            __builtin_prefetch(mxf + mxf_off(ntile, wave*2 + 1, lane), 0, 0);
        }

        // ---- fused recurrent GEMM: [hid|tops](16x512) @ [W_hh;W_sh](512x256) ----
        float nh[16];
        for (int w = 0; w < 2; ++w) {
            int nt = wave*2 + w;
            // issue mx fragment load now; consumed only after the WMMA chain
            v8f cmx = *(const v8f*)(mxf + mxf_off(tile, nt, lane));
            v8f c = {};
            for (int kt = 0; kt < 16; ++kt) {
                const _Float16* rowp = (kt < 8) ? (hrow + kt*32) : (srow + (kt-8)*32);
                v16h a = load_a_frag(rowp, hi);
                v16h b = *(const v16h*)(frags + OFF_WR + (size_t)(kt*16+nt)*FRAG_H + lane*16);
                c = wmma_f16(a, b, c);
            }
            for (int r = 0; r < 8; ++r) {
                float v = tanhf(c[r] + cmx[r]);
                nh[w*8 + r] = v;
                out[outbase + (size_t)(mBase + r)*256 + nt*16 + nn] = v;
            }
        }

        // ---- push_val = tanh(hid @ W_hs + b_hs)  (waves 0-3, N-tiles 0-3) ----
        if (wave < 4) {
            int nt = wave;
            v8f c = {};
            for (int kt = 0; kt < 8; ++kt) {
                v16h a = load_a_frag(hrow + kt*32, hi);
                v16h b = *(const v16h*)(frags + OFF_WHS + (size_t)(kt*4+nt)*FRAG_H + lane*16);
                c = wmma_f16(a, b, c);
            }
            for (int r = 0; r < 8; ++r)
                pushv[(mBase + r)*64 + nt*16 + nn] = (_Float16)tanhf(c[r] + bhs0);
        } else if (wave == 4) {
            // ---- gate logits = hid @ W_ha + b_ha, softmax over 3 ----
            v8f c = {};
            for (int kt = 0; kt < 8; ++kt) {
                v16h a = load_a_frag(hrow + kt*32, hi);
                v16h b = *(const v16h*)(frags + OFF_WHA + (size_t)kt*FRAG_H + lane*16);
                c = wmma_f16(a, b, c);
            }
            if (nn < 3)
                for (int r = 0; r < 8; ++r) logits[mBase + r][nn] = c[r] + bha0;
            if (lane < 16) {                     // same wave: DS ordering suffices
                float a0 = logits[lane][0], a1 = logits[lane][1], a2 = logits[lane][2];
                float m3 = fmaxf(a0, fmaxf(a1, a2));
                float e0 = expf(a0 - m3), e1 = expf(a1 - m3), e2 = expf(a2 - m3);
                float s = 1.0f / (e0 + e1 + e2);
                pvals[lane][0] = e0*s; pvals[lane][1] = e1*s; pvals[lane][2] = e2*s;
            }
        }
        __syncthreads();   // all reads of hid/stack done; gates & push_val published

        // ---- commit new hid ----
        for (int w = 0; w < 2; ++w) {
            int nt = wave*2 + w;
            for (int r = 0; r < 8; ++r)
                hid[(mBase + r)*256 + nt*16 + nn] = (_Float16)nh[w*8 + r];
        }

        // ---- soft stack shift: new = p_push*push + p_pop*pop + p_noop*old ----
        // packed f16 math -> v_pk_fma_f16 (2 elems/op)
        {
            const int b   = tid >> 4;
            const int sl0 = tid & 15;
            const _Float16 pps = (_Float16)pvals[b][0];
            const _Float16 pqs = (_Float16)pvals[b][1];
            const _Float16 pns = (_Float16)pvals[b][2];
            const _Float16* os = &stk[cur][(size_t)b*4096];
            _Float16*       ns = &stk[cur ^ 1][(size_t)b*4096];
            for (int j = 0; j < 4; ++j) {
                int s = sl0*4 + j;
                const _Float16* oldrow  = os + s*64;
                _Float16*       newrow  = ns + s*64;
                const _Float16* pushsrc = (s == 0) ? (pushv + b*64) : (oldrow - 64);
                const _Float16* popsrc  = (s == SSZ_ - 1) ? botv : (oldrow + 64);
                for (int dc = 0; dc < 64; dc += 8) {
                    v8h pv = *(const v8h*)(pushsrc + dc);
                    v8h ov = *(const v8h*)(oldrow + dc);
                    v8h qv = *(const v8h*)(popsrc + dc);
                    v8h rs = pps * pv + pqs * qv + pns * ov;
                    *(v8h*)(newrow + dc) = rs;
                }
            }
        }
        __syncthreads();
        cur ^= 1;
    }
}

// ---------------- hid_final gather via lens ----------------
__global__ __launch_bounds__(256) void gather_final(const int* __restrict__ lens,
                                                    float* __restrict__ out)
{
    int b = blockIdx.x, h = threadIdx.x;
    int t = lens[b] - 1;
    out[(size_t)T_*B_*HDIM_ + (size_t)b*HDIM_ + h] =
        out[((size_t)t*B_ + b)*HDIM_ + h];
}

extern "C" void kernel_launch(void* const* d_in, const int* in_sizes, int n_in,
                              void* d_out, int out_size, void* d_ws, size_t ws_size,
                              hipStream_t stream)
{
    const float* embs     = (const float*)d_in[0];
    const int*   lens     = (const int*)  d_in[1];
    const float* W_eh     = (const float*)d_in[2];
    const float* b_eh     = (const float*)d_in[3];
    const float* W_hh     = (const float*)d_in[4];
    const float* b_hh     = (const float*)d_in[5];
    const float* W_ha     = (const float*)d_in[6];
    const float* b_ha     = (const float*)d_in[7];
    const float* W_hs     = (const float*)d_in[8];
    const float* b_hs     = (const float*)d_in[9];
    const float* W_sh     = (const float*)d_in[10];
    const float* b_sh     = (const float*)d_in[11];
    const float* mem_bias = (const float*)d_in[12];
    float* out = (float*)d_out;

    _Float16* frags = (_Float16*)d_ws;
    float*    mxf   = (float*)((char*)d_ws + MX_OFF_BYTES);

    repack_weights<<<N_WEH + N_WR + N_WHS + N_WHA, 32, 0, stream>>>(
        W_eh, W_hh, W_sh, W_hs, W_ha, frags);
    precompute_mx<<<(T_ * B_) / 16, 256, 0, stream>>>(
        embs, b_eh, b_hh, b_sh, frags, mxf);
    sarnn_serial<<<B_ / 16, 256, 0, stream>>>(
        mem_bias, b_hs, b_ha, frags, mxf, out);
    gather_final<<<B_, 256, 0, stream>>>(lens, out);
}
